// MultiheadAttention_30210799960409
// MI455X (gfx1250) — matrix operand order repaired and checked by reading, not compiled, not run on hardware
//
#include <hip/hip_runtime.h>
#include <hip/hip_bf16.h>
#include <math.h>

// MI455X (gfx1250) multi-head attention forward.
//   1) QKV projection  (f32 -> f16 workspace, head-interleaved), WMMA f16
//   2) flash attention (online softmax, causal analytic), WMMA f16,
//      Q/K tiles moved with GLOBAL_LOAD_ASYNC_TO_LDS_B128 (ASYNCcnt path)
//   3) output projection (f16 ctx @ Wo + bo -> f32 d_out), WMMA f16
// B-operands stored transposed in LDS (fragment = 2x ds_load_b128);
// f32 staging uses batched b128 gathers -> cvt -> packed/scatter LDS stores.

typedef __attribute__((ext_vector_type(16))) _Float16 v16h;
typedef __attribute__((ext_vector_type(8)))  _Float16 v8h;
typedef __attribute__((ext_vector_type(4)))  _Float16 v4h;
typedef __attribute__((ext_vector_type(8)))  float    v8f;
typedef __attribute__((ext_vector_type(4)))  float    v4f;

#define E_DIM 1024
#define NHEAD 16
#define HDIM  64
#define BATCH 2
#define SEQ   2048
#define MTOT  (BATCH * SEQ)   // 4096 rows

// -------- CDNA5 async global->LDS path (guarded; fallback = VGPR staging) --
#if __has_builtin(__builtin_amdgcn_global_load_async_to_lds_b128) && \
    __has_builtin(__builtin_amdgcn_s_wait_asynccnt)
#define USE_ASYNC_LDS 1
typedef int vsi4 __attribute__((vector_size(16)));            // b128 payload
typedef __attribute__((address_space(1))) vsi4* gvsi4p;       // global src
typedef __attribute__((address_space(3))) vsi4* lvsi4p;       // LDS dst
__device__ inline gvsi4p gbl_cast(const void* p) {
  return (gvsi4p)(uintptr_t)p;                // inttoptr, per-lane global addr
}
__device__ inline lvsi4p lds_cast(void* p) {
  return (lvsi4p)(uint32_t)(uintptr_t)p;      // low 32 bits = LDS byte offset
}
#else
#define USE_ASYNC_LDS 0
#endif

// ---------------- WMMA fragment helpers (16x16x32 f16, wave32) -------------
// A fragment: 16x32 (M x K). Lane holds row M = lane&15.
// VGPR v, elem e: K = (v<4?0:16) + (lane>>4)*8 + (v&3)*2 + e   (ISA 7.12.2)
// -> two contiguous 16B chunks per lane.
__device__ inline v16h load_a_frag(const _Float16* lds, int stride) {
  const int lane = threadIdx.x & 31;
  const int half = lane >> 4;
  const int m    = lane & 15;
  v16h a;
#pragma unroll
  for (int v = 0; v < 8; ++v) {
    const int k0 = ((v < 4) ? 0 : 16) + half * 8 + ((v & 3) << 1);
    const _Float16* p = lds + m * stride + k0;
    a[2 * v]     = p[0];
    a[2 * v + 1] = p[1];
  }
  return a;
}

// B fragment: 32x16 (K x N) read from TRANSPOSED LDS: B[k][n] = lds[n*stride+k].
// Lane holds col N = lane&15; K = (v<4?0:16) + (lane>>4)*8 + (v&3)*2 + e (same as A).
__device__ inline v16h load_b_frag_T(const _Float16* lds, int stride) {
  const int lane = threadIdx.x & 31;
  const int half = lane >> 4;
  const int n    = lane & 15;
  v16h b;
#pragma unroll
  for (int v = 0; v < 8; ++v) {
    const int k0 = ((v < 4) ? 0 : 16) + half * 8 + ((v & 3) << 1);
    const _Float16* p = lds + n * stride + k0;
    b[2 * v]     = p[0];
    b[2 * v + 1] = p[1];
  }
  return b;
}

__device__ inline v8f wmma16(v16h a, v16h b, v8f c) {
  return __builtin_amdgcn_wmma_f32_16x16x32_f16(false, a, false, b,
                                                (short)0, c, false, false);
}

__device__ inline v8f vzero8() {
  v8f z = {0.f, 0.f, 0.f, 0.f, 0.f, 0.f, 0.f, 0.f};
  return z;
}

// ---------------- Kernel 1: QKV projection --------------------------------
// grid (E/64, M/128, 3), block 128 (4 waves). 128x64 tile, wave owns 128x16.
__global__ __launch_bounds__(128)
void qkv_proj_kernel(const float* __restrict__ q_in, const float* __restrict__ k_in,
                     const float* __restrict__ v_in,
                     const float* __restrict__ Wq, const float* __restrict__ bq,
                     const float* __restrict__ Wk, const float* __restrict__ bk,
                     const float* __restrict__ Wv, const float* __restrict__ bv,
                     _Float16* __restrict__ Qo, _Float16* __restrict__ Ko,
                     _Float16* __restrict__ Vo) {
  const int z = blockIdx.z;
  const float* X    = (z == 0) ? q_in : ((z == 1) ? k_in : v_in);
  const float* W    = (z == 0) ? Wq   : ((z == 1) ? Wk   : Wv);
  const float* bias = (z == 0) ? bq   : ((z == 1) ? bk   : bv);
  _Float16*    out  = (z == 0) ? Qo   : ((z == 1) ? Ko   : Vo);

  const int mb = blockIdx.y * 128;
  const int nb = blockIdx.x * 64;

  __shared__ __align__(16) _Float16 sX[128 * 40];  // 128x32, pitch 80B
  __shared__ __align__(16) _Float16 sWT[64 * 40];  // W transposed: [n][k]

  const int tid  = threadIdx.x;
  const int wave = tid >> 5;

  v8f acc[8];
#pragma unroll
  for (int i = 0; i < 8; ++i) acc[i] = vzero8();

  for (int kb = 0; kb < E_DIM; kb += 32) {
    __syncthreads();
    // ---- X: 128x32 f32, batched b128 gather -> cvt -> ds_store_b64 ----
    v4f xv[8];
#pragma unroll
    for (int i = 0; i < 8; ++i) {           // 1024 float4 chunks / 128 lanes
      const int id = tid + i * 128;
      const int r = id >> 3, c4 = (id & 7) * 4;
      xv[i] = *(const v4f*)(X + (size_t)(mb + r) * E_DIM + kb + c4);
    }
    // ---- W: 32x64 f32, batched b128 gather -> cvt -> transposed scatter ----
    v4f wv[4];
#pragma unroll
    for (int i = 0; i < 4; ++i) {           // 512 float4 chunks / 128 lanes
      const int id = tid + i * 128;
      const int r = id >> 4, c4 = (id & 15) * 4;
      wv[i] = *(const v4f*)(W + (size_t)(kb + r) * E_DIM + nb + c4);
    }
#pragma unroll
    for (int i = 0; i < 8; ++i) {
      const int id = tid + i * 128;
      const int r = id >> 3, c4 = (id & 7) * 4;
      v4h hx;
#pragma unroll
      for (int j = 0; j < 4; ++j) hx[j] = (_Float16)xv[i][j];
      *(v4h*)(&sX[r * 40 + c4]) = hx;       // 8B packed store
    }
#pragma unroll
    for (int i = 0; i < 4; ++i) {
      const int id = tid + i * 128;
      const int r = id >> 4, c4 = (id & 15) * 4;
#pragma unroll
      for (int j = 0; j < 4; ++j)
        sWT[(c4 + j) * 40 + r] = (_Float16)wv[i][j];
    }
    __syncthreads();

    const v16h bf = load_b_frag_T(sWT + (wave * 16) * 40, 40);
#pragma unroll
    for (int ms = 0; ms < 8; ++ms) {
      const v16h af = load_a_frag(sX + ms * 16 * 40, 40);
      acc[ms] = wmma16(af, bf, acc[ms]);
    }
  }

  // Epilogue: add bias, store f16 remapped to [B][H][L][HD]
  const int lane    = tid & 31;
  const int halfsel = lane >> 4;
  const int ncol    = nb + wave * 16 + (lane & 15);
  const int h       = ncol >> 6;            // / HDIM
  const int d       = ncol & 63;
  const float bias_n = bias[ncol];
#pragma unroll
  for (int ms = 0; ms < 8; ++ms) {
#pragma unroll
    for (int r = 0; r < 8; ++r) {
      const int m  = mb + ms * 16 + r + 8 * halfsel;
      const int bi = m >> 11;               // / SEQ
      const int l  = m & (SEQ - 1);
      out[(((size_t)bi * NHEAD + h) * SEQ + l) * HDIM + d] =
          (_Float16)(acc[ms][r] + bias_n);
    }
  }
}

// ---------------- Kernel 2: flash attention --------------------------------
// grid (L/64, H, B), block 128. Wave w owns query rows [qb+16w, qb+16w+16).
__global__ __launch_bounds__(128)
void flash_attn_kernel(const _Float16* __restrict__ Q, const _Float16* __restrict__ K,
                       const _Float16* __restrict__ V,
                       const unsigned char* __restrict__ kpad,
                       _Float16* __restrict__ ctx) {
  const int qb = blockIdx.x * 64;
  const int h  = blockIdx.y;
  const int b  = blockIdx.z;
  const size_t head_base = ((size_t)b * NHEAD + h) * (size_t)SEQ * HDIM;

  __shared__ __align__(16) _Float16 sQ[64 * 72];    // [q][d], pitch 144B
  __shared__ __align__(16) _Float16 sK[32 * 72];    // [key][d] (read transposed)
  __shared__ __align__(16) _Float16 sVT[64 * 40];   // V transposed: [d][key]
  __shared__ __align__(16) _Float16 sP[4][16 * 40]; // per-wave P, pitch 80B

  const int tid     = threadIdx.x;
  const int wave    = tid >> 5;
  const int lane    = tid & 31;
  const int halfsel = lane >> 4;
  const int ncol    = lane & 15;

  // ---- Q tile 64x64 f16 (8 KB) ----
#if USE_ASYNC_LDS
#pragma unroll
  for (int i = 0; i < 4; ++i) {             // 512 x 16B chunks / 128 lanes
    const int e = tid + i * 128;
    const int r = e >> 3, c8 = (e & 7) * 8;
    __builtin_amdgcn_global_load_async_to_lds_b128(
        gbl_cast(Q + head_base + (size_t)(qb + r) * HDIM + c8),
        lds_cast(&sQ[r * 72 + c8]), 0, 0);
  }
  __builtin_amdgcn_s_wait_asynccnt(0);
#else
#pragma unroll
  for (int i = 0; i < 4; ++i) {
    const int e = tid + i * 128;
    const int r = e >> 3, c8 = (e & 7) * 8;
    *(v8h*)(&sQ[r * 72 + c8]) =
        *(const v8h*)(Q + head_base + (size_t)(qb + r) * HDIM + c8);
  }
#endif
  __syncthreads();
  v16h aq[2];
  aq[0] = load_a_frag(sQ + wave * 16 * 72, 72);
  aq[1] = load_a_frag(sQ + wave * 16 * 72 + 32, 72);

  v8f o[4];
#pragma unroll
  for (int i = 0; i < 4; ++i) o[i] = vzero8();
  float m_run[8], l_run[8];
#pragma unroll
  for (int r = 0; r < 8; ++r) { m_run[r] = -1e30f; l_run[r] = 0.f; }

  const float scale = 0.125f;               // 1/sqrt(64)

  // Causal: only key tiles with kt <= qb+63 contribute.
  for (int kt = 0; kt < qb + 64; kt += 32) {
    __syncthreads();                        // all waves done reading old tiles
#if USE_ASYNC_LDS
#pragma unroll
    for (int i = 0; i < 2; ++i) {           // K tile: 256 x 16B chunks (async)
      const int e = tid + i * 128;
      const int r = e >> 3, c8 = (e & 7) * 8;
      __builtin_amdgcn_global_load_async_to_lds_b128(
          gbl_cast(K + head_base + (size_t)(kt + r) * HDIM + c8),
          lds_cast(&sK[r * 72 + c8]), 0, 0);
    }
#else
#pragma unroll
    for (int i = 0; i < 2; ++i) {
      const int e = tid + i * 128;
      const int r = e >> 3, c8 = (e & 7) * 8;
      *(v8h*)(&sK[r * 72 + c8]) =
          *(const v8h*)(K + head_base + (size_t)(kt + r) * HDIM + c8);
    }
#endif
    // V tile: batched b128 gather, then TRANSPOSED scatter into sVT[d][key]
    v8h vv[2];
#pragma unroll
    for (int i = 0; i < 2; ++i) {
      const int c   = tid + i * 128;        // 16B chunk id
      const int key = c >> 3, d0 = (c & 7) * 8;
      vv[i] = *(const v8h*)(V + head_base + (size_t)(kt + key) * HDIM + d0);
    }
#pragma unroll
    for (int i = 0; i < 2; ++i) {
      const int c   = tid + i * 128;
      const int key = c >> 3, d0 = (c & 7) * 8;
#pragma unroll
      for (int j = 0; j < 8; ++j)
        sVT[(d0 + j) * 40 + key] = vv[i][j];
    }
#if USE_ASYNC_LDS
    __builtin_amdgcn_s_wait_asynccnt(0);
#endif
    __syncthreads();

    // Prefetch next tile into caches (global_prefetch_b8)
    if (kt + 32 < qb + 64) {
      const int pr = tid >> 2;              // 32 rows covered by 128 lanes
      __builtin_prefetch(K + head_base + (size_t)(kt + 32 + pr) * HDIM, 0, 1);
      __builtin_prefetch(V + head_base + (size_t)(kt + 32 + pr) * HDIM, 0, 1);
    }

    // S = Q * K^T (16 q rows x 32 keys per wave); sK is [key][d] so the
    // transposed read gives B[d][key] with contiguous d-chunks.
    v8f s[2];
    s[0] = vzero8(); s[1] = vzero8();
#pragma unroll
    for (int ns = 0; ns < 2; ++ns)
#pragma unroll
      for (int ds = 0; ds < 2; ++ds) {
        const v16h bk = load_b_frag_T(sK + (ns * 16) * 72 + ds * 32, 72);
        s[ns] = wmma16(aq[ds], bk, s[ns]);
      }

    // scale + causal/pad mask + row max (row = r + 8*halfsel in C layout)
    float rmax[8];
#pragma unroll
    for (int r = 0; r < 8; ++r) rmax[r] = -1e30f;
#pragma unroll
    for (int ns = 0; ns < 2; ++ns) {
      const int  kg  = kt + ns * 16 + ncol;
      const bool pad = kpad[(size_t)b * SEQ + kg] != 0;
#pragma unroll
      for (int r = 0; r < 8; ++r) {
        const int qg = qb + wave * 16 + r + 8 * halfsel;
        float val = s[ns][r] * scale;
        if (pad || kg > qg) val = -1e30f;
        s[ns][r] = val;
        rmax[r]  = fmaxf(rmax[r], val);
      }
    }
#pragma unroll
    for (int r = 0; r < 8; ++r)
#pragma unroll
      for (int msk = 1; msk < 16; msk <<= 1)
        rmax[r] = fmaxf(rmax[r], __shfl_xor(rmax[r], msk, 32));

    float alpha[8], rsum[8];
#pragma unroll
    for (int r = 0; r < 8; ++r) {
      const float mnew = fmaxf(m_run[r], rmax[r]);
      alpha[r] = __expf(m_run[r] - mnew);
      m_run[r] = mnew;
      rsum[r]  = 0.f;
    }

    // P = exp(S - m); row sums; stash P (f16) for A-fragment reload
#pragma unroll
    for (int ns = 0; ns < 2; ++ns)
#pragma unroll
      for (int r = 0; r < 8; ++r) {
        const float p = __expf(s[ns][r] - m_run[r]);
        rsum[r] += p;
        sP[wave][(r + 8 * halfsel) * 40 + ns * 16 + ncol] = (_Float16)p;
      }
#pragma unroll
    for (int r = 0; r < 8; ++r) {
#pragma unroll
      for (int msk = 1; msk < 16; msk <<= 1)
        rsum[r] += __shfl_xor(rsum[r], msk, 32);
      l_run[r] = l_run[r] * alpha[r] + rsum[r];
    }

    // rescale running context (rows align with stats lane-locally)
#pragma unroll
    for (int nd = 0; nd < 4; ++nd)
#pragma unroll
      for (int r = 0; r < 8; ++r)
        o[nd][r] *= alpha[r];

    __syncthreads();                        // publish sP
    const v16h ap = load_a_frag(sP[wave], 40);
#pragma unroll
    for (int nd = 0; nd < 4; ++nd) {
      const v16h bv = load_b_frag_T(sVT + (nd * 16) * 40, 40);
      o[nd] = wmma16(ap, bv, o[nd]);
    }
  }

  // Epilogue: normalize, store ctx f16 as [B*L][E] (head-major columns)
#pragma unroll
  for (int r = 0; r < 8; ++r) {
    const float inv = 1.0f / l_run[r];
    const int   qg  = qb + wave * 16 + r + 8 * halfsel;
    const size_t row = ((size_t)b * SEQ + qg) * E_DIM + (size_t)h * HDIM;
#pragma unroll
    for (int nd = 0; nd < 4; ++nd)
      ctx[row + nd * 16 + ncol] = (_Float16)(o[nd][r] * inv);
  }
}

// ---------------- Kernel 3: output projection ------------------------------
// grid (E/64, M/128), block 128. ctx tile is f16 -> async LDS path.
__global__ __launch_bounds__(128)
void out_proj_kernel(const _Float16* __restrict__ ctx, const float* __restrict__ Wo,
                     const float* __restrict__ bo, float* __restrict__ out) {
  const int mb = blockIdx.y * 128;
  const int nb = blockIdx.x * 64;

  __shared__ __align__(16) _Float16 sX[128 * 40]; // pitch 80B
  __shared__ __align__(16) _Float16 sWT[64 * 40]; // Wo transposed: [n][k]

  const int tid  = threadIdx.x;
  const int wave = tid >> 5;

  v8f acc[8];
#pragma unroll
  for (int i = 0; i < 8; ++i) acc[i] = vzero8();

  for (int kb = 0; kb < E_DIM; kb += 32) {
    __syncthreads();
#if USE_ASYNC_LDS
#pragma unroll
    for (int i = 0; i < 4; ++i) {           // 128x32 f16 = 512 x 16B chunks
      const int e = tid + i * 128;
      const int r = e >> 2, c8 = (e & 3) * 8;
      __builtin_amdgcn_global_load_async_to_lds_b128(
          gbl_cast(ctx + (size_t)(mb + r) * E_DIM + kb + c8),
          lds_cast(&sX[r * 40 + c8]), 0, 0);
    }
#else
#pragma unroll
    for (int i = 0; i < 4; ++i) {
      const int e = tid + i * 128;
      const int r = e >> 2, c8 = (e & 3) * 8;
      *(v8h*)(&sX[r * 40 + c8]) =
          *(const v8h*)(ctx + (size_t)(mb + r) * E_DIM + kb + c8);
    }
#endif
    // W: batched b128 gather -> cvt -> transposed scatter
    v4f wv[4];
#pragma unroll
    for (int i = 0; i < 4; ++i) {
      const int id = tid + i * 128;
      const int r = id >> 4, c4 = (id & 15) * 4;
      wv[i] = *(const v4f*)(Wo + (size_t)(kb + r) * E_DIM + nb + c4);
    }
#pragma unroll
    for (int i = 0; i < 4; ++i) {
      const int id = tid + i * 128;
      const int r = id >> 4, c4 = (id & 15) * 4;
#pragma unroll
      for (int j = 0; j < 4; ++j)
        sWT[(c4 + j) * 40 + r] = (_Float16)wv[i][j];
    }
#if USE_ASYNC_LDS
    __builtin_amdgcn_s_wait_asynccnt(0);
#endif
    __syncthreads();

    const v16h bf = load_b_frag_T(sWT + (wave * 16) * 40, 40);
#pragma unroll
    for (int ms = 0; ms < 8; ++ms) {
      const v16h af = load_a_frag(sX + ms * 16 * 40, 40);
      acc[ms] = wmma16(af, bf, acc[ms]);
    }
  }

  const int lane    = tid & 31;
  const int halfsel = lane >> 4;
  const int ncol    = nb + wave * 16 + (lane & 15);
  const float bias  = bo[ncol];
#pragma unroll
  for (int ms = 0; ms < 8; ++ms)
#pragma unroll
    for (int r = 0; r < 8; ++r) {
      const int m = mb + ms * 16 + r + 8 * halfsel;
      out[(size_t)m * E_DIM + ncol] = acc[ms][r] + bias;
    }
}

// ---------------- Host launcher -------------------------------------------
extern "C" void kernel_launch(void* const* d_in, const int* in_sizes, int n_in,
                              void* d_out, int out_size, void* d_ws, size_t ws_size,
                              hipStream_t stream) {
  (void)in_sizes; (void)n_in; (void)out_size; (void)ws_size;

  const float* q_in = (const float*)d_in[0];
  const float* k_in = (const float*)d_in[1];
  const float* v_in = (const float*)d_in[2];
  const unsigned char* kpad = (const unsigned char*)d_in[3];
  // d_in[4] = attn_mask: triu(ones,k=1) causal mask, applied analytically.
  const float* Wq = (const float*)d_in[5];
  const float* bq = (const float*)d_in[6];
  const float* Wk = (const float*)d_in[7];
  const float* bk = (const float*)d_in[8];
  const float* Wv = (const float*)d_in[9];
  const float* bv = (const float*)d_in[10];
  const float* Wo = (const float*)d_in[11];
  const float* bo = (const float*)d_in[12];

  const size_t n = (size_t)MTOT * E_DIM;    // 4M elems per tensor
  _Float16* Qw = (_Float16*)d_ws;           // [B][H][L][HD] f16
  _Float16* Kw = Qw + n;
  _Float16* Vw = Kw + n;
  _Float16* Cw = Vw + n;                    // ctx [B*L][E] f16   (32 MB total)

  qkv_proj_kernel<<<dim3(E_DIM / 64, MTOT / 128, 3), 128, 0, stream>>>(
      q_in, k_in, v_in, Wq, bq, Wk, bk, Wv, bv, Qw, Kw, Vw);

  flash_attn_kernel<<<dim3(SEQ / 64, NHEAD, BATCH), 128, 0, stream>>>(
      Qw, Kw, Vw, kpad, Cw);

  out_proj_kernel<<<dim3(E_DIM / 64, MTOT / 128), 128, 0, stream>>>(
      Cw, Wo, bo, (float*)d_out);
}
